// LSTMForecaster_16776142258918
// MI455X (gfx1250) — matrix operand order repaired and checked by reading, not compiled, not run on hardware
//
#include <hip/hip_runtime.h>

// ---------------------------------------------------------------------------
// Fused 2-layer LSTM (H=50, F=4, B=4096, T=512) + final FC for gfx1250.
// One wave32 handles a 16-row batch tile; both layers scanned step-by-step,
// recurrent GEMMs done with v_wmma_f32_16x16x32_bf16, state in LDS/VGPRs.
// ---------------------------------------------------------------------------

typedef __attribute__((ext_vector_type(16))) __bf16 v16bf;
typedef __attribute__((ext_vector_type(8)))  float  v8f;

#define BATCH 4096
#define TSTEPS 512
#define HID 50
#define GATES 200     // 4*HID
#define NTILES 13     // ceil(208/16)
#define GSTRIDE 212   // gate buffer row stride (floats), padded vs 64 banks

// LDS byte layout
#define W0_BYTES   26624   // 13 ntiles * 2 ksteps * 1024B  (K=64: [x(4)|h0(50)|0])
#define W1_BYTES   53248   // 13 ntiles * 4 ksteps * 1024B  (K=128: [h0(50)|h1(50)|0])
#define BS_BYTES   848     // 212 floats (bias sum, padded)
#define OFF_W1     (W0_BYTES)
#define OFF_BS0    (OFF_W1 + W1_BYTES)           // 79872
#define OFF_BS1    (OFF_BS0 + BS_BYTES)          // 80720
#define SHARED_RO  (OFF_BS1 + BS_BYTES)          // 81568 (read-only region, from ws)
#define GBUF_BYTES (16 * GSTRIDE * 4)            // 13568
#define A0_BYTES   (16 * 72 * 2)                 // 2304  (row stride 72 halves = 144B)
#define A1_BYTES   (16 * 136 * 2)                // 4352  (row stride 136 halves = 272B)
#define HF_BYTES   (16 * 52 * 4)                 // 3328
#define WAVE_BYTES (GBUF_BYTES + A0_BYTES + A1_BYTES + HF_BYTES)   // 23552
#define SMEM_BYTES (SHARED_RO + 2 * WAVE_BYTES)  // 128672

__device__ __forceinline__ unsigned short f2bf(float x) {
  union { float f; unsigned u; } c; c.f = x;
  unsigned r = c.u + 0x7fffu + ((c.u >> 16) & 1u);
  return (unsigned short)(r >> 16);
}

// gfx1250 V_TANH_F32 (TRANS op) via builtin; verified present last round.
__device__ __forceinline__ float tanh_(float x) {
#if __has_builtin(__builtin_amdgcn_tanhf)
  return __builtin_amdgcn_tanhf(x);
#elif __has_builtin(__builtin_amdgcn_tanh_f32)
  return __builtin_amdgcn_tanh_f32(x);
#else
  float e = __expf(-2.f * x);
  return (1.f - e) / (1.f + e);
#endif
}
__device__ __forceinline__ float sigm(float x) {
#if __has_builtin(__builtin_amdgcn_tanhf) || __has_builtin(__builtin_amdgcn_tanh_f32)
  return 0.5f + 0.5f * tanh_(0.5f * x);   // one TRANS op
#else
  return 1.f / (1.f + __expf(-x));
#endif
}

__device__ __forceinline__ v16bf as_v16bf(uint4 a, uint4 b) {
  union { struct { uint4 a, b; } s; v16bf v; } u;
  u.s.a = a; u.s.b = b;
  return u.v;
}

__device__ __forceinline__ void store_tile(float* __restrict__ gbuf,
                                           const float* __restrict__ bsum,
                                           v8f acc, int n, int m, int hi) {
  const int col = n * 16 + m;
  const float bc = bsum[col];   // padded to 212, zeros past 199
#pragma unroll
  for (int r = 0; r < 8; ++r)
    gbuf[(r + hi * 8) * GSTRIDE + col] = acc[r] + bc;  // row = r + 8*hi, N = col
}

// GEMM step: gates[16,208] = A[16,32*KSTEPS] x B[32*KSTEPS,208] (+bias), via WMMA.
// N-tiles processed in pairs with two independent accumulators; the bias-add +
// LDS store of pair (np-1) is issued AFTER pair np's WMMAs so the previous
// pair's dependent VALU fills the WMMA->VALU hazard window (4 co-exec slots)
// instead of v_nop padding, and loads stay deeper in flight.
template <int KSTEPS>
__device__ __forceinline__ void lstm_mm(const char* __restrict__ wp,
                                        const char* __restrict__ abuf, int arow,
                                        float* __restrict__ gbuf,
                                        const float* __restrict__ bsum, int lane) {
  const int m  = lane & 15;
  const int hi = lane >> 4;
  v16bf af[KSTEPS];
  const char* ar = abuf + m * arow + hi * 32;   // hi selects K+16 half
#pragma unroll
  for (int kk = 0; kk < KSTEPS; ++kk)
    af[kk] = as_v16bf(*(const uint4*)(ar + kk * 64),
                      *(const uint4*)(ar + kk * 64 + 16));

  const char* bbase = wp + lane * 32;
  v8f p0, p1;   // previous pair, pending store
#pragma unroll
  for (int np = 0; np < NTILES / 2; ++np) {
    const int n0 = 2 * np, n1 = n0 + 1;
    v8f acc0 = {}, acc1 = {};
#pragma unroll
    for (int kk = 0; kk < KSTEPS; ++kk) {
      const char* b0p = bbase + (n0 * KSTEPS + kk) * 1024;
      const char* b1p = bbase + (n1 * KSTEPS + kk) * 1024;
      v16bf bf0 = as_v16bf(*(const uint4*)b0p, *(const uint4*)(b0p + 16));
      v16bf bf1 = as_v16bf(*(const uint4*)b1p, *(const uint4*)(b1p + 16));
      acc0 = __builtin_amdgcn_wmma_f32_16x16x32_bf16(
          false, af[kk], false, bf0, (short)0, acc0, false, false);
      acc1 = __builtin_amdgcn_wmma_f32_16x16x32_bf16(
          false, af[kk], false, bf1, (short)0, acc1, false, false);
    }
    if (np > 0) {   // store pair np-1 behind pair np's WMMAs
      store_tile(gbuf, bsum, p0, 2 * np - 2, m, hi);
      store_tile(gbuf, bsum, p1, 2 * np - 1, m, hi);
    }
    p0 = acc0; p1 = acc1;   // full unroll -> SSA renaming, no copies
  }
  {  // tail tile n = 12; its WMMAs cover the last pair's stores
    v8f acc = {};
#pragma unroll
    for (int kk = 0; kk < KSTEPS; ++kk) {
      const char* bp = bbase + (12 * KSTEPS + kk) * 1024;
      acc = __builtin_amdgcn_wmma_f32_16x16x32_bf16(
          false, af[kk], false, as_v16bf(*(const uint4*)bp, *(const uint4*)(bp + 16)),
          (short)0, acc, false, false);
    }
    store_tile(gbuf, bsum, p0, 10, m, hi);
    store_tile(gbuf, bsum, p1, 11, m, hi);
    store_tile(gbuf, bsum, acc, 12, m, hi);
  }
}

// Gate nonlinearity: 800 (m,j) pairs over 32 lanes; c kept in VGPRs.
__device__ __forceinline__ void gates_ew(const float* __restrict__ gbuf,
                                         float* __restrict__ creg, int lane,
                                         char* d0, int s0, char* d1, int s1,
                                         float* hfin) {
#pragma unroll
  for (int it = 0; it < 25; ++it) {
    int idx = it * 32 + lane;
    int m = idx & 15;
    int j = idx >> 4;               // 0..49
    const float* gm = gbuf + m * GSTRIDE;
    float gi = gm[j];
    float gf = gm[50 + j];
    float gg = gm[100 + j];
    float go = gm[150 + j];
    float c = sigm(gf) * creg[it] + sigm(gi) * tanh_(gg);
    float h = sigm(go) * tanh_(c);
    creg[it] = c;
    unsigned short hb = f2bf(h);
    *(unsigned short*)(d0 + m * s0 + j * 2) = hb;
    if (d1) *(unsigned short*)(d1 + m * s1 + j * 2) = hb;
    if (hfin) hfin[m * 52 + j] = h;
  }
}

// ---------------------------------------------------------------------------
// Pack kernel: swizzle f32 weights into bf16 B-operand blocks + bias sums in ws.
// Block (ntile n, kstep kk): 32 lanes x 16 halves; lane l holds
// B[K = 32kk + 16*(l/16) + hh][N = 16n + (l%16)] at byte l*32 + hh*2.
// B[k][ncol] = Wcat[ncol][k].
// ---------------------------------------------------------------------------
#define PACK_W0 13312           // 13*2*512 halves
#define PACK_W1 26624           // 13*4*512 halves
#define PACK_TOT (PACK_W0 + PACK_W1 + 424)

__global__ void lstm_pack_kernel(const float* __restrict__ Wih0, const float* __restrict__ Whh0,
                                 const float* __restrict__ bih0, const float* __restrict__ bhh0,
                                 const float* __restrict__ Wih1, const float* __restrict__ Whh1,
                                 const float* __restrict__ bih1, const float* __restrict__ bhh1,
                                 char* __restrict__ ws) {
  int tid = blockIdx.x * 256 + threadIdx.x;
  unsigned short* w0 = (unsigned short*)ws;
  unsigned short* w1 = (unsigned short*)(ws + OFF_W1);
  float* bs0 = (float*)(ws + OFF_BS0);
  float* bs1 = (float*)(ws + OFF_BS1);
  if (tid < PACK_W0) {
    int block = tid >> 9, e = tid & 511;
    int n = block >> 1, kk = block & 1;
    int l = e >> 4, hh = e & 15;
    int ncol = n * 16 + (l & 15);
    int k = kk * 32 + ((l >> 4) << 4) + hh;       // K within cat vector [x|h0]
    float v = 0.f;
    if (ncol < GATES) {
      if (k < 4)       v = Wih0[ncol * 4 + k];
      else if (k < 54) v = Whh0[ncol * 50 + (k - 4)];
    }
    w0[tid] = f2bf(v);
  } else if (tid < PACK_W0 + PACK_W1) {
    int t2 = tid - PACK_W0;
    int block = t2 >> 9, e = t2 & 511;
    int n = block >> 2, kk = block & 3;
    int l = e >> 4, hh = e & 15;
    int ncol = n * 16 + (l & 15);
    int k = kk * 32 + ((l >> 4) << 4) + hh;       // K within cat vector [h0|h1]
    float v = 0.f;
    if (ncol < GATES) {
      if (k < 50)       v = Wih1[ncol * 50 + k];
      else if (k < 100) v = Whh1[ncol * 50 + (k - 50)];
    }
    w1[t2] = f2bf(v);
  } else if (tid < PACK_TOT) {
    int k = tid - (PACK_W0 + PACK_W1);
    if (k < 212) bs0[k] = (k < GATES) ? bih0[k] + bhh0[k] : 0.f;
    else {
      int k2 = k - 212;
      bs1[k2] = (k2 < GATES) ? bih1[k2] + bhh1[k2] : 0.f;
    }
  }
}

// ---------------------------------------------------------------------------
// Main fused kernel: 2 waves/WG, one 16-row batch tile per wave.
// ---------------------------------------------------------------------------
__global__ void __launch_bounds__(64) lstm_fused_kernel(const float* __restrict__ x,
                                                        const float* __restrict__ fcW,
                                                        const float* __restrict__ fcb,
                                                        const char* __restrict__ wsg,
                                                        float* __restrict__ out) {
  extern __shared__ char smem[];
  const int tid = threadIdx.x;
  const int lane = tid & 31;
  const int wv = tid >> 5;

  // Stage packed weights + biases into LDS (shared by both waves).
  for (int i = tid; i < SHARED_RO / 16; i += 64)
    ((float4*)smem)[i] = ((const float4*)wsg)[i];

  char* W0p = smem;
  char* W1p = smem + OFF_W1;
  const float* bs0 = (const float*)(smem + OFF_BS0);
  const float* bs1 = (const float*)(smem + OFF_BS1);

  char* wbase = smem + SHARED_RO + wv * WAVE_BYTES;
  float* gbuf = (float*)wbase;
  char* A0 = wbase + GBUF_BYTES;                  // [16 rows x 72 halves] bf16
  char* A1 = A0 + A0_BYTES;                       // [16 rows x 136 halves] bf16 (contiguous)
  float* hfin = (float*)(A1 + A1_BYTES);          // [16 x 52] f32

  // Zero A0+A1 (h_{-1}=0 and K zero-padding).
  for (int i = lane; i < (A0_BYTES + A1_BYTES) / 4; i += 32)
    ((unsigned int*)A0)[i] = 0u;
  __syncthreads();

  const int b0 = blockIdx.x * 32 + wv * 16;
  float c0[25], c1[25];
#pragma unroll
  for (int i = 0; i < 25; ++i) { c0[i] = 0.f; c1[i] = 0.f; }

  const float4* xg = (const float4*)x;            // x is [B, T, 4] f32 -> float4 per (b,t)

  // Software-pipeline the x stream: xv always holds x[., t] at loop top.
  float4 xv = make_float4(0.f, 0.f, 0.f, 0.f);
  if (lane < 16) xv = xg[(size_t)(b0 + lane) * TSTEPS];

  for (int t = 0; t < TSTEPS; ++t) {
    // Feed x_t into layer-0 A buffer at K=0..3 (bf16), then prefetch x_{t+1}.
    if (lane < 16) {
      uint2 q;
      q.x = (unsigned)f2bf(xv.x) | ((unsigned)f2bf(xv.y) << 16);
      q.y = (unsigned)f2bf(xv.z) | ((unsigned)f2bf(xv.w) << 16);
      *(uint2*)(A0 + lane * 144) = q;
      if (t + 1 < TSTEPS)
        xv = xg[(size_t)(b0 + lane) * TSTEPS + t + 1];   // latency hidden by step
    }
    // Layer 0: gates = [x_t | h0] (K=64) x W0cat^T  -> 13 x 2 WMMA
    lstm_mm<2>(W0p, A0, 144, gbuf, bs0, lane);
    // h0_t -> A0 K=4..53 (next step), A1 K=0..49 (layer-1 input this step)
    gates_ew(gbuf, c0, lane, A0 + 8, 144, A1, 272, nullptr);
    // Layer 1: gates = [h0_t | h1] (K=128) x W1cat^T -> 13 x 4 WMMA
    lstm_mm<4>(W1p, A1, 272, gbuf, bs1, lane);
    // h1_t -> A1 K=50..99; capture final hidden at t = T-1
    gates_ew(gbuf, c1, lane, A1 + 100, 272, nullptr, 0,
             (t == TSTEPS - 1) ? hfin : nullptr);
  }

  // Final FC: out[b, f] = hT[b, :] . fcW[f, :] + fcb[f]   (2 outputs per lane)
  const int m = lane & 15;
  const int fb = lane >> 4;
  for (int f = fb; f < 4; f += 2) {
    float s = fcb[f];
#pragma unroll 10
    for (int j = 0; j < HID; ++j)
      s += hfin[m * 52 + j] * fcW[f * HID + j];
    out[(size_t)(b0 + m) * 4 + f] = s;
  }
}

// ---------------------------------------------------------------------------
extern "C" void kernel_launch(void* const* d_in, const int* in_sizes, int n_in,
                              void* d_out, int out_size, void* d_ws, size_t ws_size,
                              hipStream_t stream) {
  const float* x    = (const float*)d_in[0];
  const float* Wih0 = (const float*)d_in[1];
  const float* Whh0 = (const float*)d_in[2];
  const float* bih0 = (const float*)d_in[3];
  const float* bhh0 = (const float*)d_in[4];
  const float* Wih1 = (const float*)d_in[5];
  const float* Whh1 = (const float*)d_in[6];
  const float* bih1 = (const float*)d_in[7];
  const float* bhh1 = (const float*)d_in[8];
  const float* fcW  = (const float*)d_in[9];
  const float* fcb  = (const float*)d_in[10];
  char* ws = (char*)d_ws;
  float* out = (float*)d_out;

  lstm_pack_kernel<<<(PACK_TOT + 255) / 256, 256, 0, stream>>>(
      Wih0, Whh0, bih0, bhh0, Wih1, Whh1, bih1, bhh1, ws);

  hipFuncSetAttribute(reinterpret_cast<const void*>(lstm_fused_kernel),
                      hipFuncAttributeMaxDynamicSharedMemorySize, SMEM_BYTES);
  lstm_fused_kernel<<<BATCH / 32, 64, SMEM_BYTES, stream>>>(x, fcW, fcb, ws, out);
}
